// BiLevelRoutingAttention_86096914415836
// MI455X (gfx1250) — matrix operand order repaired and checked by reading, not compiled, not run on hardware
//
#include <hip/hip_runtime.h>
#include <hip/hip_bf16.h>

// ---------------- types ----------------
typedef _Float16 h16v __attribute__((ext_vector_type(16)));
typedef _Float16 h8v  __attribute__((ext_vector_type(8)));
typedef _Float16 h4v  __attribute__((ext_vector_type(4)));
typedef float    f8v  __attribute__((ext_vector_type(8)));
typedef float    f4v  __attribute__((ext_vector_type(4)));

// ---------------- problem constants ----------------
#define TT    4
#define BBATCH 2
#define NW    32
#define WS    128
#define CC    256
#define NH    8
#define HD    32
#define TOPK  4
#define ROWS  (TT*BBATCH*NW*WS)   // 32768
#define QKVLD 768                 // q|k|v packed per row
#define SCALE 0.17677669529663687f // 1/sqrt(32)

// window-partition mapping: (t,b,w,s) -> flat channel-base index into x
__device__ __forceinline__ long xbase_of(int t, int b, int w, int s) {
  int a = w >> 4, bw = (w >> 2) & 3, cw = w & 3;   // wt=2, wh=4, ww=4
  int i = s >> 6, j = (s >> 3) & 7, kk = s & 7;    // Lt/wt=2, Lh/wh=8, Lw/ww=8
  int lt = a * 2 + i, lh = bw * 8 + j, lw = cw * 8 + kk;
  long idx = ((((long)t * BBATCH + b) * 4 + lt) * 32 + lh) * 32 + lw;
  return idx * CC;
}

// ---------------- WMMA fragment helpers (ISA 7.12.2 layouts) ----------------
// A 16x32 f16: lane m = lane&15; lanes 0-15 hold K 0-7,16-23; lanes 16-31 hold K 8-15,24-31
__device__ __forceinline__ h16v load_a_f16(const _Float16* rowp) {
  int k0 = (threadIdx.x & 16) ? 8 : 0;
  h8v lo = *(const h8v*)(rowp + k0);
  h8v hi = *(const h8v*)(rowp + 16 + k0);
  return __builtin_shufflevector(lo, hi, 0,1,2,3,4,5,6,7,8,9,10,11,12,13,14,15);
}
__device__ __forceinline__ h16v load_a_f32(const float* rowp) {
  int k0 = (threadIdx.x & 16) ? 8 : 0;
  f4v x0 = *(const f4v*)(rowp + k0);
  f4v x1 = *(const f4v*)(rowp + k0 + 4);
  f4v x2 = *(const f4v*)(rowp + 16 + k0);
  f4v x3 = *(const f4v*)(rowp + 16 + k0 + 4);
  h4v h0 = __builtin_convertvector(x0, h4v);
  h4v h1 = __builtin_convertvector(x1, h4v);
  h4v h2 = __builtin_convertvector(x2, h4v);
  h4v h3 = __builtin_convertvector(x3, h4v);
  h8v lo = __builtin_shufflevector(h0, h1, 0,1,2,3,4,5,6,7);
  h8v hi = __builtin_shufflevector(h2, h3, 0,1,2,3,4,5,6,7);
  return __builtin_shufflevector(lo, hi, 0,1,2,3,4,5,6,7,8,9,10,11,12,13,14,15);
}
// B 32x16 f16 from a row-major [N][K] matrix: colp points at row n's K=0.
// lanes 0-15 hold K 0-7,16-23; lanes 16-31 hold K 8-15,24-31.
__device__ __forceinline__ h16v load_b_f16(const _Float16* colp) {
  int k0 = (threadIdx.x & 16) ? 8 : 0;
  h8v lo = *(const h8v*)(colp + k0);
  h8v hi = *(const h8v*)(colp + 16 + k0);
  return __builtin_shufflevector(lo, hi, 0,1,2,3,4,5,6,7,8,9,10,11,12,13,14,15);
}
__device__ __forceinline__ f8v wmma_f16(h16v a, h16v b, f8v c) {
  return __builtin_amdgcn_wmma_f32_16x16x32_f16(false, a, false, b, (short)0, c, false, false);
}

// ---------------- k1: weight f32 -> f16 ----------------
__global__ void k_wprep(const float* wqkv, const float* wproj,
                        _Float16* wq16, _Float16* wp16) {
  int i = blockIdx.x * blockDim.x + threadIdx.x;
  if (i < 768 * 256) wq16[i] = (_Float16)wqkv[i];
  if (i < 256 * 256) wp16[i] = (_Float16)wproj[i];
}

// ---------------- k2: region = mean over (T, ws) ----------------
__global__ void k_region(const float* __restrict__ x, float* __restrict__ region) {
  int bw = blockIdx.x;            // b*NW + w
  int b = bw / NW, w = bw % NW;
  int ch = threadIdx.x;
  float acc = 0.f;
  for (int t = 0; t < TT; ++t)
    for (int s = 0; s < WS; ++s)
      acc += x[xbase_of(t, b, w, s) + ch];
  region[(long)bw * CC + ch] = acc * (1.f / (TT * WS));
}

// ---------------- k3: scores + top-4 routing ----------------
__global__ void k_route(const float* __restrict__ region, int* __restrict__ idxout) {
  __shared__ float sc[NW][NW];
  int b = blockIdx.x, tid = threadIdx.x;
  for (int p = tid; p < NW * NW; p += blockDim.x) {
    int i = p / NW, j = p % NW;
    const float* ri = region + ((long)b * NW + i) * CC;
    const float* rj = region + ((long)b * NW + j) * CC;
    float d = 0.f;
    for (int c = 0; c < CC; ++c) d += ri[c] * rj[c];
    sc[i][j] = d * SCALE;
  }
  __syncthreads();
  if (tid < NW) {
    unsigned mask = 0;
    for (int r = 0; r < TOPK; ++r) {
      float best = -3.4e38f; int bj = 0;
      for (int j = 0; j < NW; ++j)
        if (!((mask >> j) & 1u) && sc[tid][j] > best) { best = sc[tid][j]; bj = j; }
      mask |= (1u << bj);
      idxout[((long)b * NW + tid) * TOPK + r] = bj;
    }
  }
}

// ---------------- k4: QKV GEMM (WMMA): qkv[row][768] = x_win[row] @ Wqkv^T ----------------
__global__ void k_qkv(const float* __restrict__ x, const _Float16* __restrict__ w16,
                      _Float16* __restrict__ qkv) {
  int blk = blockIdx.x;           // (t,b,w), 256 blocks
  int w = blk % NW, tb = blk / NW, b = tb % BBATCH, t = tb / BBATCH;
  int wave = threadIdx.x >> 5, lane = threadIdx.x & 31, m = lane & 15;
  int srow = wave * 16 + m;
  const float* arow = x + xbase_of(t, b, w, srow);
  h16v a[8];
#pragma unroll
  for (int kt = 0; kt < 8; ++kt) a[kt] = load_a_f32(arow + kt * 32);
  long rowbase = (long)blk * WS + wave * 16;
  int mh = (lane & 16) ? 8 : 0;
  for (int nt = 0; nt < 48; ++nt) {
    const _Float16* bp = w16 + (long)(nt * 16 + m) * CC;
    if (nt + 1 < 48) __builtin_prefetch(w16 + (long)((nt + 1) * 16 + m) * CC, 0, 0);
    f8v c = {};
#pragma unroll
    for (int kt = 0; kt < 8; ++kt) c = wmma_f16(a[kt], load_b_f16(bp + kt * 32), c);
#pragma unroll
    for (int r = 0; r < 8; ++r)
      qkv[(rowbase + r + mh) * QKVLD + nt * 16 + m] = (_Float16)c[r];
  }
}

// ---------------- k5: in-place LayerNorm + LIF spike on each 256-ch segment ----------------
__global__ void k_lnlif(_Float16* __restrict__ qkv,
                        const float* g_q, const float* b_q,
                        const float* g_k, const float* b_k,
                        const float* g_v, const float* b_v) {
  __shared__ float red[512];
  long row = blockIdx.x;
  int ch = threadIdx.x;
  for (int seg = 0; seg < 3; ++seg) {
    _Float16* p = qkv + row * QKVLD + seg * CC;
    float v = (float)p[ch];
    red[ch] = v; red[256 + ch] = v * v;
    __syncthreads();
    for (int off = 128; off > 0; off >>= 1) {
      if (ch < off) { red[ch] += red[ch + off]; red[256 + ch] += red[256 + ch + off]; }
      __syncthreads();
    }
    float mean = red[0] * (1.f / CC);
    float var  = red[256] * (1.f / CC) - mean * mean;
    const float* g = (seg == 0) ? g_q : (seg == 1) ? g_k : g_v;
    const float* bi = (seg == 0) ? b_q : (seg == 1) ? b_k : b_v;
    float y = (v - mean) * rsqrtf(var + 1e-5f) * g[ch] + bi[ch];
    // LIF: spike = (y/tau - vth >= 0), tau=2, vth=0.5  <=>  y >= 1
    p[ch] = (_Float16)((y >= 1.0f) ? 1.0f : 0.0f);
    __syncthreads();
  }
}

// ---------------- k6: routed linear attention (WMMA) ----------------
__global__ void k_attn(const _Float16* __restrict__ qkv, const int* __restrict__ idx,
                       _Float16* __restrict__ att) {
  __shared__ union {
    struct { _Float16 k[32][CC]; _Float16 v[32][CC]; } st;                       // 32 KB
    struct { _Float16 kv[NH][32][32]; float ksum[NH][32]; float den[NH][16]; } p2;
  } sm;
  int blk = blockIdx.x;           // (t,b,w)
  int w = blk % NW, tb = blk / NW, b = tb % BBATCH;
  int head = threadIdx.x >> 5, lane = threadIdx.x & 31, e = lane & 15;
  int mh = (lane & 16) ? 8 : 0;
  long rowbase = (long)blk * WS;
  long tbbase  = (long)tb * NW * WS;

  f8v c[2][2] = {{{}, {}}, {{}, {}}};
  float ksump[2] = {0.f, 0.f};

  for (int j = 0; j < TOPK; ++j) {
    int w2 = idx[((long)b * NW + w) * TOPK + j];
    long rb2 = tbbase + (long)w2 * WS;
    for (int chnk = 0; chnk < 4; ++chnk) {
      // cooperative stage of 32 rows of k (seg1) and v (seg2) into LDS
      int r = threadIdx.x >> 3;
      int ch0 = (threadIdx.x & 7) * 32;
      const _Float16* kp = qkv + (rb2 + chnk * 32 + r) * QKVLD + CC + ch0;
      const _Float16* vp = qkv + (rb2 + chnk * 32 + r) * QKVLD + 2 * CC + ch0;
#pragma unroll
      for (int q8 = 0; q8 < 4; ++q8) {
        *(h8v*)&sm.st.k[r][ch0 + q8 * 8] = *(const h8v*)(kp + q8 * 8);
        *(h8v*)&sm.st.v[r][ch0 + q8 * 8] = *(const h8v*)(vp + q8 * 8);
      }
      __syncthreads();
      // A = k^T (rows = d), B = v (cols = e), K-dim = 32 staged s-values
      h16v afr[2], bfr[2];
#pragma unroll
      for (int dt = 0; dt < 2; ++dt) {
        int d = head * HD + dt * 16 + (lane & 15);
        float ks = 0.f;
#pragma unroll
        for (int hh = 0; hh < 16; ++hh) {
          int s = (hh < 8 ? hh : hh + 8) + ((lane & 16) ? 8 : 0);
          _Float16 kvv = sm.st.k[s][d];
          afr[dt][hh] = kvv;
          ks += (float)kvv;
        }
        ksump[dt] += ks;
      }
#pragma unroll
      for (int et = 0; et < 2; ++et) {
        int ecol = head * HD + et * 16 + e;
#pragma unroll
        for (int hh = 0; hh < 16; ++hh) {
          int s = (hh < 8 ? hh : hh + 8) + ((lane & 16) ? 8 : 0);
          bfr[et][hh] = sm.st.v[s][ecol];
        }
      }
#pragma unroll
      for (int dt = 0; dt < 2; ++dt)
#pragma unroll
        for (int et = 0; et < 2; ++et)
          c[dt][et] = wmma_f16(afr[dt], bfr[et], c[dt][et]);
      __syncthreads();
    }
  }

  // publish per-head kv (scaled) and k_sum into LDS (staging region is dead now)
#pragma unroll
  for (int dt = 0; dt < 2; ++dt) {
#pragma unroll
    for (int et = 0; et < 2; ++et)
#pragma unroll
      for (int r = 0; r < 8; ++r)
        sm.p2.kv[head][dt * 16 + r + mh][et * 16 + e] = (_Float16)(c[dt][et][r] * SCALE);
    float ks = ksump[dt] + __shfl_xor(ksump[dt], 16);
    if (lane < 16) sm.p2.ksum[head][dt * 16 + lane] = ks * SCALE;
  }
  __syncthreads();

  // phase 2: out = q @ kv, den = q . ksum, out /= (|den| + 1e-4)
  for (int mt = 0; mt < 8; ++mt) {
    const _Float16* qrow = qkv + (rowbase + mt * 16 + (lane & 15)) * QKVLD + head * HD;
    h16v aq = load_a_f16(qrow);
    float dp = 0.f;
#pragma unroll
    for (int hh = 0; hh < 16; ++hh) {
      int d = (hh < 8 ? hh : hh + 8) + ((lane & 16) ? 8 : 0);
      dp += (float)aq[hh] * sm.p2.ksum[head][d];
    }
    dp += __shfl_xor(dp, 16);
    if (lane < 16) sm.p2.den[head][lane] = dp;
    __syncthreads();
    f8v o[2] = {{}, {}};
#pragma unroll
    for (int et = 0; et < 2; ++et) {
      h16v bkv;
#pragma unroll
      for (int hh = 0; hh < 16; ++hh) {
        int d = (hh < 8 ? hh : hh + 8) + ((lane & 16) ? 8 : 0);
        bkv[hh] = sm.p2.kv[head][d][et * 16 + e];
      }
      o[et] = wmma_f16(aq, bkv, o[et]);
    }
#pragma unroll
    for (int et = 0; et < 2; ++et)
#pragma unroll
      for (int r = 0; r < 8; ++r) {
        int m = r + mh;
        float den = sm.p2.den[head][m];
        float val = o[et][r] / (fabsf(den) + 1e-4f);
        att[(rowbase + mt * 16 + m) * CC + head * HD + et * 16 + e] = (_Float16)val;
      }
    __syncthreads();
  }
}

// ---------------- k7: output projection (WMMA) + bias ----------------
__global__ void k_proj(const _Float16* __restrict__ att, const _Float16* __restrict__ wp16,
                       const float* __restrict__ bias, _Float16* __restrict__ att2) {
  int blk = blockIdx.x;           // 256 blocks x 128 rows
  int wave = threadIdx.x >> 5, lane = threadIdx.x & 31, m = lane & 15;
  long row0 = (long)blk * 128 + wave * 16;
  const _Float16* arow = att + (row0 + m) * CC;
  h16v a[8];
#pragma unroll
  for (int kt = 0; kt < 8; ++kt) a[kt] = load_a_f16(arow + kt * 32);
  int mh = (lane & 16) ? 8 : 0;
  for (int nt = 0; nt < 16; ++nt) {
    const _Float16* bp = wp16 + (long)(nt * 16 + m) * CC;
    if (nt + 1 < 16) __builtin_prefetch(wp16 + (long)((nt + 1) * 16 + m) * CC, 0, 0);
    f8v c = {};
#pragma unroll
    for (int kt = 0; kt < 8; ++kt) c = wmma_f16(a[kt], load_b_f16(bp + kt * 32), c);
    float bv = bias[nt * 16 + m];
#pragma unroll
    for (int r = 0; r < 8; ++r)
      att2[(row0 + r + mh) * CC + nt * 16 + m] = (_Float16)(c[r] + bv);
  }
}

// ---------------- k8: final LayerNorm + window-merge scatter ----------------
__global__ void k_lnout(const _Float16* __restrict__ att2, const float* __restrict__ g_o,
                        const float* __restrict__ b_o, float* __restrict__ out) {
  __shared__ float red[512];
  long row = blockIdx.x;
  int ch = threadIdx.x;
  float v = (float)att2[row * CC + ch];
  red[ch] = v; red[256 + ch] = v * v;
  __syncthreads();
  for (int off = 128; off > 0; off >>= 1) {
    if (ch < off) { red[ch] += red[ch + off]; red[256 + ch] += red[256 + ch + off]; }
    __syncthreads();
  }
  float mean = red[0] * (1.f / CC);
  float var  = red[256] * (1.f / CC) - mean * mean;
  float y = (v - mean) * rsqrtf(var + 1e-5f) * g_o[ch] + b_o[ch];
  int s = (int)(row & 127), w = (int)((row >> 7) & 31);
  int b = (int)((row >> 12) & 1), t = (int)(row >> 13);
  out[xbase_of(t, b, w, s) + ch] = y;
}

// ---------------- launch ----------------
extern "C" void kernel_launch(void* const* d_in, const int* in_sizes, int n_in,
                              void* d_out, int out_size, void* d_ws, size_t ws_size,
                              hipStream_t stream) {
  const float* x      = (const float*)d_in[0];
  const float* Wqkv   = (const float*)d_in[1];
  const float* g_q    = (const float*)d_in[2];
  const float* b_q    = (const float*)d_in[3];
  const float* g_k    = (const float*)d_in[4];
  const float* b_k    = (const float*)d_in[5];
  const float* g_v    = (const float*)d_in[6];
  const float* b_v    = (const float*)d_in[7];
  const float* Wproj  = (const float*)d_in[8];
  const float* b_proj = (const float*)d_in[9];
  const float* g_o    = (const float*)d_in[10];
  const float* b_o    = (const float*)d_in[11];
  float* out = (float*)d_out;

  char* ws = (char*)d_ws;
  int*      idx    = (int*)(ws + 0);                 //   1 KB
  float*    region = (float*)(ws + 4096);            //  64 KB
  _Float16* wq16   = (_Float16*)(ws + 69632);        // 384 KB
  _Float16* wp16   = (_Float16*)(ws + 462848);       // 128 KB
  _Float16* qkv    = (_Float16*)(ws + (1u << 20));   //  48 MB  [row][768]
  _Float16* att    = (_Float16*)(ws + 52428800u);    //  16 MB  [row][256]
  _Float16* att2   = qkv;                            // reuse qkv region after attention

  k_wprep <<<768, 256, 0, stream>>>(Wqkv, Wproj, wq16, wp16);
  k_region<<<BBATCH * NW, 256, 0, stream>>>(x, region);
  k_route <<<BBATCH, 256, 0, stream>>>(region, idx);
  k_qkv   <<<TT * BBATCH * NW, 256, 0, stream>>>(x, wq16, qkv);
  k_lnlif <<<ROWS, 256, 0, stream>>>(qkv, g_q, b_q, g_k, b_k, g_v, b_v);
  k_attn  <<<TT * BBATCH * NW, 256, 0, stream>>>(qkv, idx, att);
  k_proj  <<<ROWS / 128, 256, 0, stream>>>(att, wp16, b_proj, att2);
  k_lnout <<<ROWS, 256, 0, stream>>>(att2, g_o, b_o, out);
}
